// CoFix3DDecoder_87273735454761
// MI455X (gfx1250) — matrix operand order repaired and checked by reading, not compiled, not run on hardware
//
#include <hip/hip_runtime.h>
#include <hip/hip_bf16.h>
#include <stdint.h>

// ---------------- types ----------------
typedef _Float16 v16h __attribute__((ext_vector_type(16)));
typedef _Float16 v8h  __attribute__((ext_vector_type(8)));
typedef float    v8f  __attribute__((ext_vector_type(8)));
typedef unsigned int u32x4 __attribute__((ext_vector_type(4)));
typedef int      i32x8 __attribute__((ext_vector_type(8)));
typedef int      i32x4 __attribute__((ext_vector_type(4)));

#define HH 360
#define WW 360
#define CIN 128
#define NC 10
#define NP 200
#define HWP (HH*WW)
#define CAND_CAP 4096
#define NBINS 4096

#if defined(__has_builtin)
#if __has_builtin(__builtin_amdgcn_sched_barrier)
#define SCHED_FENCE() __builtin_amdgcn_sched_barrier(0)
#endif
#endif
#ifndef SCHED_FENCE
#define SCHED_FENCE() ((void)0)
#endif

// ---------------- TDM: 1-D tile load into LDS (amdgpu-toolchain 6-arg builtin) ----------------
// D# built per CDNA5 ISA ch.10.8: group0 = {count/lds_addr/global_addr/type},
// group1 = {data_size, tensor_dim0/1, tile_dim0/1, strides}. data_size=1 (2B elems).
__device__ __forceinline__ void tdm_load_1d(void* lds_ptr, const void* gptr, unsigned nelem) {
  unsigned lds_off = (unsigned)(uintptr_t)lds_ptr;          // low 32 bits = LDS offset
  unsigned long long ga = (unsigned long long)(uintptr_t)gptr;
  u32x4 g0;
  g0[0] = 1u;                                               // count=1, gather off
  g0[1] = lds_off;                                          // lds_addr
  g0[2] = (unsigned)(ga & 0xFFFFFFFFull);                   // global_addr[31:0]
  g0[3] = (unsigned)((ga >> 32) & 0x1FFFFFFull) | (2u << 30); // global_addr[56:32] | type=2
  i32x8 g1;
  unsigned w0 = (1u << 16);                                 // data_size=1 (2 bytes)
  unsigned w1 = (nelem & 0xFFFFu) << 16;                    // tensor_dim0[15:0]
  unsigned w2 = ((nelem >> 16) & 0xFFFFu) | (1u << 16);     // tensor_dim0[31:16] | tensor_dim1=1
  unsigned w3 = (nelem & 0xFFFFu) << 16;                    // tile_dim0
  g1[0] = (int)w0; g1[1] = (int)w1; g1[2] = (int)w2; g1[3] = (int)w3;
  g1[4] = 1;                // tile_dim1=1, tile_dim2=0
  g1[5] = (int)nelem;       // tensor_dim0_stride[31:0]
  g1[6] = 0; g1[7] = 0;
  i32x4 g2; g2[0] = 1; g2[1] = 1; g2[2] = 0; g2[3] = 0;     // tensor_dim2/3 = 1
  i32x4 g3; g3[0] = 0; g3[1] = 0; g3[2] = 0; g3[3] = 0;
  i32x8 g4; g4[0] = 0; g4[1] = 0; g4[2] = 0; g4[3] = 0;
  g4[4] = 0; g4[5] = 0; g4[6] = 0; g4[7] = 0;
  __builtin_amdgcn_tensor_load_to_lds(g0, g1, g2, g3, g4, 0);
}

// ---------------- pack feat NCHW f32 -> NHWC f16 ----------------
__global__ __launch_bounds__(256) void pack_feat_kernel(const float* __restrict__ feat,
                                                        _Float16* __restrict__ featH) {
  __shared__ _Float16 tile[64 * 132];   // [pixel][ch], padded stride 132
  int bid = blockIdx.x;
  int xt = bid % 6;
  int y  = (bid / 6) % HH;
  int b  = bid / (6 * HH);
  int x0 = xt * 64;
  int t = threadIdx.x;
  // read phase: (c, x) coalesced along x
  for (int i = 0; i < 32; ++i) {
    int l = i * 256 + t;            // over 128*64
    int c = l >> 6, x = l & 63;
    float v = 0.f;
    if (x0 + x < WW) v = feat[((size_t)(b * CIN + c)) * HWP + (size_t)y * WW + x0 + x];
    tile[x * 132 + c] = (_Float16)v;
  }
  __syncthreads();
  // write phase: 8B chunks, channel-contiguous
  for (int i = 0; i < 8; ++i) {
    int chunk = i * 256 + t;        // over 64*32
    int px = chunk >> 5;
    int off = (chunk & 31) << 2;
    if (x0 + px < WW) {
      _Float16* d = &featH[(((size_t)(b * HH + y)) * WW + x0 + px) * CIN + off];
      const _Float16* s = &tile[px * 132 + off];
      *(unsigned long long*)d = *(const unsigned long long*)s;
    }
  }
}

// ---------------- pack weights -> [tap][co][ci] f16 ----------------
__global__ __launch_bounds__(256) void pack_w1_kernel(const float* __restrict__ w1,
                                                      _Float16* __restrict__ w1p) {
  int i = blockIdx.x * 256 + threadIdx.x;   // over 9*128*128
  if (i >= 9 * 128 * 128) return;
  int ci = i & 127;
  int co = (i >> 7) & 127;
  int tap = i >> 14;
  w1p[i] = (_Float16)w1[((size_t)(co * 128 + ci)) * 9 + tap];
}
__global__ __launch_bounds__(256) void pack_w2_kernel(const float* __restrict__ w2,
                                                      _Float16* __restrict__ w2p) {
  int i = blockIdx.x * 256 + threadIdx.x;   // over 9*16*128
  if (i >= 9 * 16 * 128) return;
  int ci = i & 127;
  int co = (i >> 7) & 15;
  int tap = i >> 11;
  _Float16 v = (_Float16)0.f;
  if (co < NC) v = (_Float16)w2[((size_t)(co * 128 + ci)) * 9 + tap];
  w2p[i] = v;
}

// ---------------- conv1: 128->128 3x3 + bias + relu, implicit GEMM via WMMA ----------------
__global__ __launch_bounds__(256) void conv1_kernel(const _Float16* __restrict__ featH,
                                                    const _Float16* __restrict__ w1p,
                                                    const float* __restrict__ b1,
                                                    _Float16* __restrict__ y1) {
  extern __shared__ _Float16 smem[];
  _Float16* Alds = smem;                  // 3*130*128 halfs (reused as store tile later)
  _Float16* Wlds = smem + 3 * 130 * CIN;  // 128*128 halfs

  int bid = blockIdx.x;
  int xt = bid % 3;
  int y  = (bid / 3) % HH;
  int b  = bid / (3 * HH);
  int x0 = xt * 128;
  int tid = threadIdx.x;
  int lane = tid & 31;
  int wv = tid >> 5;

  // stage A halo: rows y-1..y+1, cols x0-1..x0+128, all channels (f16, NHWC)
  for (int i = tid; i < 3 * 2080; i += 256) {
    int row = i / 2080;
    int rem = i % 2080;
    int col = rem >> 4;
    int ch  = (rem & 15) << 3;
    int gy = y + row - 1;
    int gx = x0 + col - 1;
    v8h v = {};
    if ((unsigned)gy < HH && (unsigned)gx < WW)
      v = *(const v8h*)&featH[(((size_t)(b * HH + gy)) * WW + gx) * CIN + ch];
    *(v8h*)&Alds[(row * 130 + col) * CIN + ch] = v;
  }

  v8f acc[8];
  v8f zero = {0.f, 0.f, 0.f, 0.f, 0.f, 0.f, 0.f, 0.f};
#pragma unroll
  for (int nt = 0; nt < 8; ++nt) acc[nt] = zero;

  for (int tap = 0; tap < 9; ++tap) {
    __syncthreads();                       // prior compute done before W overwrite
    if (wv == 0) {
      tdm_load_1d(Wlds, w1p + (size_t)tap * 16384, 16384);   // 128co x 128ci tile
      __builtin_amdgcn_s_wait_tensorcnt(0);
    }
    __syncthreads();
    int row = tap / 3;
    int dx  = tap % 3 - 1;
    int m   = lane & 15;
    int col = wv * 16 + m + 1 + dx;
    int kh  = (lane >> 4) << 4;
    int n   = lane & 15;
#pragma unroll
    for (int kc = 0; kc < 4; ++kc) {
      // preload A fragment
      const _Float16* ap = &Alds[(row * 130 + col) * CIN + kc * 32 + kh];
      v8h lo = *(const v8h*)ap;
      v8h hi = *(const v8h*)(ap + 8);
      v16h a;
#pragma unroll
      for (int j = 0; j < 8; ++j) { a[j] = lo[j]; a[j + 8] = hi[j]; }
      // preload ALL 8 B fragments
      int kb = kc * 32 + ((lane >> 4) << 4);
      v16h bf[8];
#pragma unroll
      for (int nt = 0; nt < 8; ++nt)
        bf[nt] = *(const v16h*)&Wlds[(nt * 16 + n) * CIN + kb];
      // fence: keep the scheduler from sinking the ds_loads into the WMMA chain
      SCHED_FENCE();
#pragma unroll
      for (int nt = 0; nt < 8; ++nt)
        acc[nt] = __builtin_amdgcn_wmma_f32_16x16x32_f16(false, a, false, bf[nt],
                                                         (short)0, acc[nt], false, false);
      SCHED_FENCE();
    }
  }

  // ---- store: bias+relu -> f16 tile in LDS (reuse Alds) -> coalesced 16B stores ----
  __syncthreads();
  _Float16* Slds = Alds;                   // 128 px * 128 ch halfs = 32 KB
#pragma unroll
  for (int nt = 0; nt < 8; ++nt) {
    int n = nt * 16 + (lane & 15);
    float bias = b1[n];
#pragma unroll
    for (int r = 0; r < 8; ++r) {
      int M = r + ((lane >> 4) << 3);
      float v = acc[nt][r] + bias;
      v = v > 0.f ? v : 0.f;
      Slds[(wv * 16 + M) * CIN + n] = (_Float16)v;
    }
  }
  __syncthreads();
  for (int i = tid; i < 128 * 16; i += 256) {   // 128 px * 16 chunks of 8 halfs
    int px = i >> 4;
    int ch = (i & 15) << 3;
    int x = x0 + px;
    if (x < WW)
      *(v8h*)&y1[(((size_t)(b * HH + y)) * WW + x) * CIN + ch] =
          *(const v8h*)&Slds[px * CIN + ch];
  }
}

// ---------------- conv2: 128->10 (padded 16) 3x3 + bias + sigmoid ----------------
__global__ __launch_bounds__(256) void conv2_kernel(const _Float16* __restrict__ y1,
                                                    const _Float16* __restrict__ w2p,
                                                    const float* __restrict__ b2,
                                                    float* __restrict__ hm) {
  extern __shared__ _Float16 smem[];
  _Float16* Alds = smem;                   // 3*130*128 (reused for store tile later)
  _Float16* Wlds = smem + 3 * 130 * CIN;   // 9*16*128

  int bid = blockIdx.x;
  int xt = bid % 3;
  int y  = (bid / 3) % HH;
  int b  = bid / (3 * HH);
  int x0 = xt * 128;
  int tid = threadIdx.x;
  int lane = tid & 31;
  int wv = tid >> 5;

  for (int i = tid; i < 3 * 2080; i += 256) {
    int row = i / 2080;
    int rem = i % 2080;
    int col = rem >> 4;
    int ch  = (rem & 15) << 3;
    int gy = y + row - 1;
    int gx = x0 + col - 1;
    v8h v = {};
    if ((unsigned)gy < HH && (unsigned)gx < WW)
      v = *(const v8h*)&y1[(((size_t)(b * HH + gy)) * WW + gx) * CIN + ch];
    *(v8h*)&Alds[(row * 130 + col) * CIN + ch] = v;
  }
  if (wv == 0) {
    tdm_load_1d(Wlds, w2p, 9 * 16 * 128);     // all taps at once (36 KB)
    __builtin_amdgcn_s_wait_tensorcnt(0);
  }
  __syncthreads();

  v8f acc = {0.f, 0.f, 0.f, 0.f, 0.f, 0.f, 0.f, 0.f};
  int n = lane & 15;
  for (int tap = 0; tap < 9; ++tap) {
    int row = tap / 3;
    int dx  = tap % 3 - 1;
    int m   = lane & 15;
    int col = wv * 16 + m + 1 + dx;
    int kh  = (lane >> 4) << 4;
    // preload 4 A + 4 B fragments, then 4 WMMAs
    v16h af[4], bfr[4];
#pragma unroll
    for (int kc = 0; kc < 4; ++kc) {
      const _Float16* ap = &Alds[(row * 130 + col) * CIN + kc * 32 + kh];
      v8h lo = *(const v8h*)ap;
      v8h hi = *(const v8h*)(ap + 8);
#pragma unroll
      for (int j = 0; j < 8; ++j) { af[kc][j] = lo[j]; af[kc][j + 8] = hi[j]; }
      int kb = kc * 32 + ((lane >> 4) << 4);
      bfr[kc] = *(const v16h*)&Wlds[(tap * 16 + n) * CIN + kb];
    }
    SCHED_FENCE();
#pragma unroll
    for (int kc = 0; kc < 4; ++kc)
      acc = __builtin_amdgcn_wmma_f32_16x16x32_f16(false, af[kc], false, bfr[kc],
                                                   (short)0, acc, false, false);
    SCHED_FENCE();
  }

  // ---- store: bias+sigmoid -> f32 tile in LDS -> coalesced per-plane stores ----
  __syncthreads();
  float* Sf = (float*)Alds;                 // 16 ch * 128 px floats = 8 KB
  {
    int c = lane & 15;
    float bias = b2[c < NC ? c : NC - 1];
#pragma unroll
    for (int r = 0; r < 8; ++r) {
      int M = r + ((lane >> 4) << 3);
      float v = acc[r] + bias;
      v = 1.f / (1.f + expf(-v));
      Sf[c * 128 + (wv * 16 + M)] = v;
    }
  }
  __syncthreads();
  for (int i = tid; i < NC * 128; i += 256) {
    int c = i >> 7;
    int px = i & 127;
    int x = x0 + px;
    if (x < WW)
      hm[(((size_t)(b * NC + c)) * HH + y) * WW + x] = Sf[c * 128 + px];
  }
}

// ---------------- 3x3 local-max NMS mask ----------------
__global__ __launch_bounds__(256) void nms_kernel(const float* __restrict__ hm,
                                                  float* __restrict__ heat_out) {
  size_t i = (size_t)blockIdx.x * 256 + threadIdx.x;
  if (i >= (size_t)4 * NC * HWP) return;
  int x = (int)(i % WW);
  int y = (int)((i / WW) % HH);
  float v = hm[i];
  float mx = v;
  for (int dy = -1; dy <= 1; ++dy)
    for (int dx = -1; dx <= 1; ++dx) {
      int ny = y + dy, nx = x + dx;
      if ((unsigned)ny < HH && (unsigned)nx < WW)
        mx = fmaxf(mx, hm[i + (size_t)dy * WW + dx]);
    }
  heat_out[i] = (mx == v) ? v : 0.f;
}

// ---------------- top-k: histogram -> threshold -> compact -> sort ----------------
__global__ __launch_bounds__(256) void hist_kernel(const float* __restrict__ heat,
                                                   int* __restrict__ hist) {
  size_t i = (size_t)blockIdx.x * 256 + threadIdx.x;
  if (i >= (size_t)4 * NC * HWP) return;
  float v = heat[i];
  if (v > 0.f) {
    int b = (int)(i / ((size_t)NC * HWP));
    int bin = (int)(v * (float)NBINS);
    if (bin > NBINS - 1) bin = NBINS - 1;
    if (bin < 0) bin = 0;
    atomicAdd(&hist[b * NBINS + bin], 1);
  }
}

__global__ void thresh_kernel(const int* __restrict__ hist, float* __restrict__ thr) {
  int b = threadIdx.x;
  if (b >= 4) return;
  int cum = 0;
  float t = 0.f;
  for (int bin = NBINS - 1; bin >= 0; --bin) {
    cum += hist[b * NBINS + bin];
    if (cum >= NP) { t = (float)bin / (float)NBINS; break; }
  }
  thr[b] = t;
}

__global__ __launch_bounds__(256) void compact_kernel(const float* __restrict__ heat,
                                                      const float* __restrict__ thr,
                                                      int* __restrict__ cnt,
                                                      float* __restrict__ candS,
                                                      int* __restrict__ candI) {
  size_t i = (size_t)blockIdx.x * 256 + threadIdx.x;
  if (i >= (size_t)4 * NC * HWP) return;
  int b = (int)(i / ((size_t)NC * HWP));
  int flat = (int)(i - (size_t)b * NC * HWP);
  float v = heat[i];
  if (v > 0.f && v >= thr[b]) {
    int pos = atomicAdd(&cnt[b], 1);
    if (pos < CAND_CAP) {
      candS[b * CAND_CAP + pos] = v;
      candI[b * CAND_CAP + pos] = flat;
    }
  }
}

__global__ __launch_bounds__(256) void topk_sort_kernel(const float* __restrict__ candS,
                                                        const int* __restrict__ candI,
                                                        const int* __restrict__ cnt,
                                                        int* __restrict__ topI) {
  __shared__ float s[CAND_CAP];
  __shared__ int   d[CAND_CAP];
  int b = blockIdx.x;
  int n = cnt[b]; if (n > CAND_CAP) n = CAND_CAP;
  for (int i = threadIdx.x; i < CAND_CAP; i += 256) {
    if (i < n) { s[i] = candS[b * CAND_CAP + i]; d[i] = candI[b * CAND_CAP + i]; }
    else       { s[i] = -1e30f; d[i] = 0x7fffffff; }
  }
  __syncthreads();
  for (int k = 2; k <= CAND_CAP; k <<= 1) {
    for (int j = k >> 1; j > 0; j >>= 1) {
      for (int i = threadIdx.x; i < CAND_CAP; i += 256) {
        int ixj = i ^ j;
        if (ixj > i) {
          float si = s[i], sj = s[ixj];
          int di = d[i], dj = d[ixj];
          bool iGreater = (si > sj) || (si == sj && di < dj);
          bool up = ((i & k) == 0);             // descending blocks
          if (up ? !iGreater : iGreater) {
            s[i] = sj; s[ixj] = si; d[i] = dj; d[ixj] = di;
          }
        }
      }
      __syncthreads();
    }
  }
  for (int i = threadIdx.x; i < NP; i += 256) {
    int idx = d[i];
    if (idx == 0x7fffffff) idx = 0;
    topI[b * NP + i] = idx;
  }
}

// ---------------- final decode: gather + class enc + sine embed + MLPs ----------------
__global__ __launch_bounds__(256) void decode_kernel(const float* __restrict__ feat,
                                                     const float* __restrict__ w_cls,
                                                     const float* __restrict__ b_cls,
                                                     const float* __restrict__ pw1,
                                                     const float* __restrict__ pb1,
                                                     const float* __restrict__ g1,
                                                     const float* __restrict__ be1,
                                                     const float* __restrict__ pw2,
                                                     const float* __restrict__ pb2,
                                                     const float* __restrict__ g2,
                                                     const float* __restrict__ be2,
                                                     const int* __restrict__ topI,
                                                     float* __restrict__ out) {
  __shared__ float pe[256];
  __shared__ float h[128];
  __shared__ float stat[2];
  int blk = blockIdx.x;
  int b = blk / NP, p = blk % NP;
  int t = threadIdx.x;
  int idx = topI[b * NP + p];
  if (idx < 0 || idx >= NC * HWP) idx = 0;
  int pos = idx % HWP;
  int cls = idx / HWP;
  float xs = (float)(pos % WW) / (float)WW;
  float ys = (float)(pos / WW) / (float)HH;

  // sine embed: first 128 = y-embed, next 128 = x-embed
  {
    float e = (t < 128) ? ys : xs;
    int k = t & 127;
    int j = k >> 1;
    float dimt = powf(10000.f, (float)j * (1.f / 64.f));
    float val = e * 6.283185307179586f / dimt;
    pe[t] = (k & 1) ? cosf(val) : sinf(val);
  }
  __syncthreads();

  // MLP1 256->128
  if (t < 128) {
    float sum = pb1[t];
    for (int k = 0; k < 256; ++k) sum += pe[k] * pw1[t * 256 + k];
    h[t] = sum;
  }
  __syncthreads();
  if (t == 0) {
    float m = 0.f; for (int i = 0; i < 128; ++i) m += h[i]; m *= (1.f / 128.f);
    float v = 0.f; for (int i = 0; i < 128; ++i) { float dd = h[i] - m; v += dd * dd; }
    v *= (1.f / 128.f);
    stat[0] = m; stat[1] = rsqrtf(v + 1e-5f);
  }
  __syncthreads();
  if (t < 128) {
    float v = (h[t] - stat[0]) * stat[1] * g1[t] + be1[t];
    pe[t] = v > 0.f ? v : 0.f;            // reuse pe[0..127] as ln1 output
  }
  __syncthreads();

  // MLP2 128->128
  if (t < 128) {
    float sum = pb2[t];
    for (int k = 0; k < 128; ++k) sum += pe[k] * pw2[t * 128 + k];
    h[t] = sum;
  }
  __syncthreads();
  if (t == 0) {
    float m = 0.f; for (int i = 0; i < 128; ++i) m += h[i]; m *= (1.f / 128.f);
    float v = 0.f; for (int i = 0; i < 128; ++i) { float dd = h[i] - m; v += dd * dd; }
    v *= (1.f / 128.f);
    stat[0] = m; stat[1] = rsqrtf(v + 1e-5f);
  }
  __syncthreads();
  if (t < 128) {
    float v = (h[t] - stat[0]) * stat[1] * g2[t] + be2[t];
    v = v > 0.f ? v : 0.f;
    float q = feat[((size_t)(b * CIN + t)) * HWP + pos] + w_cls[t * NC + cls] + b_cls[t];
    out[((size_t)(b * NP + p)) * CIN + t] = q + v;
  }
}

// ---------------- host launcher ----------------
extern "C" void kernel_launch(void* const* d_in, const int* in_sizes, int n_in,
                              void* d_out, int out_size, void* d_ws, size_t ws_size,
                              hipStream_t stream) {
  (void)in_sizes; (void)n_in; (void)out_size; (void)ws_size;
  const float* feat  = (const float*)d_in[0];
  const float* w1    = (const float*)d_in[1];
  const float* b1    = (const float*)d_in[2];
  const float* w2    = (const float*)d_in[3];
  const float* b2    = (const float*)d_in[4];
  const float* w_cls = (const float*)d_in[5];
  const float* b_cls = (const float*)d_in[6];
  const float* pw1   = (const float*)d_in[7];
  const float* pb1   = (const float*)d_in[8];
  const float* g1    = (const float*)d_in[9];
  const float* be1   = (const float*)d_in[10];
  const float* pw2   = (const float*)d_in[11];
  const float* pb2   = (const float*)d_in[12];
  const float* g2    = (const float*)d_in[13];
  const float* be2   = (const float*)d_in[14];

  char* ws = (char*)d_ws;
  const size_t featHalfs = (size_t)4 * HH * WW * CIN;
  size_t off = 0;
  _Float16* featH = (_Float16*)(ws + off); off += featHalfs * 2;               // NHWC f16
  _Float16* y1    = (_Float16*)(ws + off); off += featHalfs * 2;               // conv1 out
  _Float16* w1p   = (_Float16*)(ws + off); off += (size_t)9 * 128 * 128 * 2;
  _Float16* w2p   = (_Float16*)(ws + off); off += (size_t)9 * 16 * 128 * 2;
  float*    hm    = (float*)(ws + off);    off += (size_t)4 * NC * HWP * 4;    // pre-NMS heatmap
  int*      hist  = (int*)(ws + off);      off += (size_t)4 * NBINS * 4;
  int*      cnt   = (int*)(ws + off);      off += 256;
  float*    thr   = (float*)(ws + off);    off += 256;
  float*    candS = (float*)(ws + off);    off += (size_t)4 * CAND_CAP * 4;
  int*      candI = (int*)(ws + off);      off += (size_t)4 * CAND_CAP * 4;
  int*      topI  = (int*)(ws + off);      off += (size_t)4 * NP * 4;

  float* outQ = (float*)d_out;                         // [4,200,128]
  float* heat = (float*)d_out + (size_t)4 * NP * CIN;  // [4,10,360,360]

  // zero histogram / counters / thresholds each call (graph-capturable)
  (void)hipMemsetAsync(hist, 0, (size_t)4 * NBINS * 4 + 512, stream);

  pack_feat_kernel<<<4 * HH * 6, 256, 0, stream>>>(feat, featH);
  pack_w1_kernel<<<(9 * 128 * 128 + 255) / 256, 256, 0, stream>>>(w1, w1p);
  pack_w2_kernel<<<(9 * 16 * 128 + 255) / 256, 256, 0, stream>>>(w2, w2p);

  size_t lds1 = (size_t)(3 * 130 * CIN + 128 * CIN) * 2;   // 132,608 B
  conv1_kernel<<<4 * HH * 3, 256, lds1, stream>>>(featH, w1p, b1, y1);

  size_t lds2 = (size_t)(3 * 130 * CIN + 9 * 16 * CIN) * 2; // 136,704 B
  conv2_kernel<<<4 * HH * 3, 256, lds2, stream>>>(y1, w2p, b2, hm);

  int heatBlocks = (int)(((size_t)4 * NC * HWP + 255) / 256);
  nms_kernel<<<heatBlocks, 256, 0, stream>>>(hm, heat);
  hist_kernel<<<heatBlocks, 256, 0, stream>>>(heat, hist);
  thresh_kernel<<<1, 32, 0, stream>>>(hist, thr);
  compact_kernel<<<heatBlocks, 256, 0, stream>>>(heat, thr, cnt, candS, candI);
  topk_sort_kernel<<<4, 256, 0, stream>>>(candS, candI, cnt, topI);
  decode_kernel<<<4 * NP, 256, 0, stream>>>(feat, w_cls, b_cls, pw1, pb1, g1, be1,
                                            pw2, pb2, g2, be2, topI, outQ);
}
